// MultiHeadAttention_58136677318961
// MI455X (gfx1250) — matrix operand order repaired and checked by reading, not compiled, not run on hardware
//
#include <hip/hip_runtime.h>
#include <hip/hip_bf16.h>

// ---------------------------------------------------------------------------
// MHA forward for MI455X (gfx1250):
//   - bf16 WMMA (v_wmma_f32_16x16x32_bf16) for all matmuls, f32 accumulate
//   - GEMM tiles: global_load_async_to_lds_b128 ping-pong (ASYNCcnt)
//   - attention K/V tiles: Tensor Data Mover (tensor_load_to_lds, TENSORcnt)
// ---------------------------------------------------------------------------

typedef __attribute__((ext_vector_type(16))) __bf16 v16bf;
typedef __attribute__((ext_vector_type(8)))  float  v8f;
typedef __attribute__((ext_vector_type(4)))  unsigned su4;
typedef __attribute__((ext_vector_type(8)))  unsigned su8;

union BF16x16 { v16bf v; __bf16 e[16]; };
union F32x8   { v8f  v; float  e[8];  };

#define S_LEN  2048
#define DMODEL 1024
#define NHEAD  16
#define DHEAD  64
#define BATCH  2

__device__ __forceinline__ v8f wmma_bf16(v16bf a, v16bf b, v8f c) {
  // D = A(16x32 bf16) * B(32x16 bf16) + C(16x16 f32)
  return __builtin_amdgcn_wmma_f32_16x16x32_bf16(
      /*neg_a=*/false, a, /*neg_b=*/false, b,
      /*c_mod=*/(short)0, c, /*reuse_a=*/false, /*reuse_b=*/false);
}

// Async global -> LDS copy, 16 bytes per lane, tracked by ASYNCcnt.
__device__ __forceinline__ void async_ld_b128(void* lds, const void* g) {
  unsigned           ldsa = (unsigned)(unsigned long long)lds;
  unsigned long long ga   = (unsigned long long)g;
  asm volatile("global_load_async_to_lds_b128 %0, %1, off"
               :: "v"(ldsa), "v"(ga) : "memory");
}
#define WAIT_ASYNC(n) asm volatile("s_wait_asynccnt " #n ::: "memory")

// Tensor Data Mover: 1-D contiguous tile (nelem 2-byte elements) -> LDS.
// D# group0: count=1 | lds_addr | global_addr[56:0] | type=2.
// D# group1: workgroup_mask=0, data_size=1 (2B), tensor_dim0, tile_dim0,
//            tensor_dim0_stride. tile_dim1/2 = 0 (unused) => 1-D transfer.
__device__ __forceinline__ void tdm_load_1d(const void* g, void* lds,
                                            unsigned nelem) {
  unsigned long long ga = (unsigned long long)g;
  unsigned ldsa = (unsigned)(unsigned long long)lds;
  const unsigned td0 = (unsigned)S_LEN * DHEAD;   // tensor length (elements)
  su4 g0;
  g0[0] = 1u;                                    // count=1 (valid descriptor)
  g0[1] = ldsa;                                  // lds_addr (bytes)
  g0[2] = (unsigned)ga;                          // global_addr[31:0]
  g0[3] = ((unsigned)(ga >> 32) & 0x01FFFFFFu)   // global_addr[56:32]
          | (2u << 30);                          // type=2 ("image")
  su8 g1;
  g1[0] = 0x00010000u;                           // data_size=1 -> 2 bytes
  g1[1] = (td0 & 0xFFFFu) << 16;                 // tensor_dim0[15:0] @ [31:16]
  g1[2] = (td0 >> 16) & 0xFFFFu;                 // tensor_dim0[31:16] @ [15:0]
  g1[3] = (nelem & 0xFFFFu) << 16;               // tile_dim0 @ [31:16]
  g1[4] = 0u;                                    // tile_dim1=0, tile_dim2=0
  g1[5] = td0;                                   // tensor_dim0_stride[31:0]
  g1[6] = 0u;
  g1[7] = 0u;
  asm volatile("tensor_load_to_lds %0, %1" :: "s"(g0), "s"(g1) : "memory");
}

// ------------------------- f32 -> bf16 convert ------------------------------
__global__ void cvt_f32_to_bf16(const float4* __restrict__ src,
                                __bf16* __restrict__ dst, int n4) {
  int i = blockIdx.x * blockDim.x + threadIdx.x;
  if (i < n4) {
    float4 f = src[i];
    dst[4 * i + 0] = (__bf16)f.x;
    dst[4 * i + 1] = (__bf16)f.y;
    dst[4 * i + 2] = (__bf16)f.z;
    dst[4 * i + 3] = (__bf16)f.w;
  }
}

// ------------------------- GEMM: out = A @ W^T ------------------------------
// A: (M x 1024) bf16 row-major.  W: (1024 x 1024) bf16 row-major.
// mode 0: store bf16 into (B,H,S,dh) layout (projection kernels)
// mode 1: store f32 row-major (final fc -> d_out)
__global__ void __launch_bounds__(256)
gemm_xwt(const __bf16* __restrict__ A, const __bf16* __restrict__ W,
         void* __restrict__ outp, int mode) {
  __shared__ __bf16 As[2][128 * 32];   // ping-pong A tiles
  __shared__ __bf16 Bs[2][64 * 32];    // ping-pong W tiles

  const int t    = threadIdx.x;
  const int lane = t & 31;
  const int w    = t >> 5;        // wave 0..7
  const int wm   = w & 3;         // 4 waves along M
  const int wn   = w >> 2;        // 2 waves along N
  const int hi   = lane >> 4;
  const int ln   = lane & 15;
  const int m0   = blockIdx.x * 128;
  const int n0   = blockIdx.y * 64;

  const int rowA = t >> 1, segA = t & 1;   // A tile: 2 x b128 per thread
  const int rowB = t >> 2, segB = t & 3;   // W tile: 1 x b128 per thread

  v8f zero = {0.f, 0.f, 0.f, 0.f, 0.f, 0.f, 0.f, 0.f};
  v8f acc[2][2] = {{zero, zero}, {zero, zero}};

  // prologue: issue tile 0
  {
    const __bf16* pa = A + (size_t)(m0 + rowA) * DMODEL + segA * 16;
    async_ld_b128(&As[0][rowA * 32 + segA * 16],     pa);
    async_ld_b128(&As[0][rowA * 32 + segA * 16 + 8], pa + 8);
    async_ld_b128(&Bs[0][rowB * 32 + segB * 8],
                  W + (size_t)(n0 + rowB) * DMODEL + segB * 8);
  }

  const int NK = DMODEL / 32;
  for (int j = 0; j < NK; ++j) {
    if (j + 1 < NK) {   // prefetch next tile, keep 3 copies in flight
      int kk = (j + 1) * 32;
      const __bf16* pa = A + (size_t)(m0 + rowA) * DMODEL + kk + segA * 16;
      __bf16* as = As[(j + 1) & 1];
      async_ld_b128(&as[rowA * 32 + segA * 16],     pa);
      async_ld_b128(&as[rowA * 32 + segA * 16 + 8], pa + 8);
      async_ld_b128(&Bs[(j + 1) & 1][rowB * 32 + segB * 8],
                    W + (size_t)(n0 + rowB) * DMODEL + kk + segB * 8);
      WAIT_ASYNC(3);    // group j complete; group j+1 may remain in flight
    } else {
      WAIT_ASYNC(0);
    }
    __syncthreads();    // tile j visible to all waves

    const __bf16* as = As[j & 1];
    const __bf16* bs = Bs[j & 1];
    BF16x16 af[2], bf[2];
    for (int fm = 0; fm < 2; ++fm) {
      int mrow = wm * 32 + fm * 16 + ln;
      for (int i = 0; i < 16; ++i) {
        int k = i + hi * 16;
        af[fm].e[i] = as[mrow * 32 + k];
      }
    }
    for (int fn = 0; fn < 2; ++fn) {
      int nrow = wn * 32 + fn * 16 + ln;
      for (int i = 0; i < 16; ++i)
        bf[fn].e[i] = bs[nrow * 32 + (i + hi * 16)];
    }
    for (int fm = 0; fm < 2; ++fm)
      for (int fn = 0; fn < 2; ++fn)
        acc[fm][fn] = wmma_bf16(af[fm].v, bf[fn].v, acc[fm][fn]);

    __syncthreads();    // everyone done reading tile j before it is rewritten
  }

  for (int fm = 0; fm < 2; ++fm)
    for (int fn = 0; fn < 2; ++fn) {
      F32x8 c; c.v = acc[fm][fn];
      for (int r = 0; r < 8; ++r) {
        int m = m0 + wm * 32 + fm * 16 + r + hi * 8;
        int n = n0 + wn * 32 + fn * 16 + ln;
        if (mode == 0) {
          int b = m >> 11, s = m & (S_LEN - 1);
          int h = n >> 6,  d = n & (DHEAD - 1);
          ((__bf16*)outp)[(((size_t)(b * NHEAD + h) * S_LEN + s) << 6) + d] =
              (__bf16)c.e[r];
        } else {
          ((float*)outp)[(size_t)m * DMODEL + n] = c.e[r];
        }
      }
    }
}

// ------------------------- attention (flash-style) --------------------------
// qp/kp/vp: (B*H, S, dh) bf16.  ao: (B*S, D) bf16 (heads concatenated).
// K/V tiles (4KB contiguous each) are DMA'd by the Tensor Data Mover: wave 0
// issues one tensor_load_to_lds per tile and pipelines with TENSORcnt.
__global__ void __launch_bounds__(256)
attn_fwd(const __bf16* __restrict__ qp, const __bf16* __restrict__ kp,
         const __bf16* __restrict__ vp, __bf16* __restrict__ ao) {
  __shared__ __bf16 kt[2][32 * 64];    // ping-pong K tiles
  __shared__ __bf16 vt[2][32 * 64];    // ping-pong V tiles
  __shared__ __bf16 pbuf[8][16 * 32];  // per-wave P re-layout buffer

  const int t    = threadIdx.x;
  const int lane = t & 31;
  const int w    = t >> 5;
  const int hi   = lane >> 4;
  const int ln   = lane & 15;
  const int bh   = blockIdx.y;                    // b*H + h
  const size_t base = (size_t)bh * S_LEN * DHEAD;
  const int q0   = blockIdx.x * 128 + w * 16;     // 16 query rows per wave

  // Q fragments: dh=64 => two K=32 A-fragments, loaded once
  BF16x16 aq[2];
  for (int ks = 0; ks < 2; ++ks)
    for (int i = 0; i < 16; ++i) {
      int d = ks * 32 + i + hi * 16;
      aq[ks].e[i] = qp[base + (size_t)(q0 + ln) * DHEAD + d];
    }

  v8f zero = {0.f, 0.f, 0.f, 0.f, 0.f, 0.f, 0.f, 0.f};
  v8f o[4] = {zero, zero, zero, zero};            // 16x64 f32 accumulator
  float mrow[8], lrow[8];
  for (int r = 0; r < 8; ++r) { mrow[r] = -1e30f; lrow[r] = 0.f; }

  // prologue: TDM-issue K/V tile 0 (wave 0 only; TDM ignores EXEC)
  if (w == 0) {
    tdm_load_1d(kp + base, &kt[0][0], 32 * DHEAD);
    tdm_load_1d(vp + base, &vt[0][0], 32 * DHEAD);
  }

  const int NKB = S_LEN / 32;
  for (int kb = 0; kb < NKB; ++kb) {
    if (w == 0) {
      if (kb + 1 < NKB) {   // prefetch next K/V block via TDM
        const size_t src = base + (size_t)(kb + 1) * 32 * DHEAD;
        tdm_load_1d(kp + src, &kt[(kb + 1) & 1][0], 32 * DHEAD);
        tdm_load_1d(vp + src, &vt[(kb + 1) & 1][0], 32 * DHEAD);
        __builtin_amdgcn_s_wait_tensorcnt(2);  // tile kb landed; kb+1 in flight
      } else {
        __builtin_amdgcn_s_wait_tensorcnt(0);
      }
    }
    __syncthreads();        // wave 0 signals only after TDM data arrived

    const __bf16* ktc = kt[kb & 1];
    const __bf16* vtc = vt[kb & 1];

    // scores: S(16 x 32) = Q(16x64) . K^T, two 16x16 C-frags
    F32x8 s0, s1;
    {
      BF16x16 bk0, bk1;
      for (int nb = 0; nb < 2; ++nb) {
        for (int i = 0; i < 16; ++i) {
          bk0.e[i] = ktc[(nb * 16 + ln) * 64 + (i + hi * 16)];
          bk1.e[i] = ktc[(nb * 16 + ln) * 64 + 32 + (i + hi * 16)];
        }
        v8f c = zero;
        c = wmma_bf16(aq[0].v, bk0.v, c);
        c = wmma_bf16(aq[1].v, bk1.v, c);
        if (nb == 0) s0.v = c; else s1.v = c;
      }
    }

    // online softmax (per row m = r + 8*hi; reductions inside 16-lane halves)
    float alpha[8];
    for (int r = 0; r < 8; ++r) {
      float a0 = s0.e[r] * 0.125f;                // 1/sqrt(64)
      float a1 = s1.e[r] * 0.125f;
      float cur = fmaxf(a0, a1);
      for (int msk = 1; msk < 16; msk <<= 1)
        cur = fmaxf(cur, __shfl_xor(cur, msk, 32));
      float mnew = fmaxf(mrow[r], cur);
      float p0 = __expf(a0 - mnew);
      float p1 = __expf(a1 - mnew);
      float rs = p0 + p1;
      for (int msk = 1; msk < 16; msk <<= 1)
        rs += __shfl_xor(rs, msk, 32);
      alpha[r] = __expf(mrow[r] - mnew);
      lrow[r]  = lrow[r] * alpha[r] + rs;
      mrow[r]  = mnew;
      int m = r + hi * 8;                         // C-frag row -> P row
      pbuf[w][m * 32 + ln]      = (__bf16)p0;
      pbuf[w][m * 32 + 16 + ln] = (__bf16)p1;
    }
    for (int f = 0; f < 4; ++f) {
      F32x8 of; of.v = o[f];
      for (int r = 0; r < 8; ++r) of.e[r] *= alpha[r];
      o[f] = of.v;
    }

    // P re-layout (C-layout -> A-fragment) via per-wave LDS buffer
    BF16x16 ap;
    for (int i = 0; i < 16; ++i) {
      int kcol = i + hi * 16;
      ap.e[i] = pbuf[w][ln * 32 + kcol];
    }
    // O += P(16x32) . V(32x64): four B-fragments over dh
    for (int f = 0; f < 4; ++f) {
      BF16x16 bv;
      for (int i = 0; i < 16; ++i)
        bv.e[i] = vtc[(i + hi * 16) * 64 + f * 16 + ln];
      o[f] = wmma_bf16(ap.v, bv.v, o[f]);
    }

    __syncthreads();   // everyone done with tile kb before TDM rewrites it
  }

  // normalize and store heads-concatenated: ao[(b*S+s)*D + h*64 + d]
  int b = bh >> 4, h = bh & 15;
  for (int f = 0; f < 4; ++f) {
    F32x8 of; of.v = o[f];
    for (int r = 0; r < 8; ++r) {
      int m = r + hi * 8;
      int s = q0 + m;
      float val = of.e[r] / lrow[r];
      ao[((size_t)(b * S_LEN + s)) * DMODEL + h * DHEAD + f * 16 + ln] = (__bf16)val;
    }
  }
}

// ---------------------------------------------------------------------------
extern "C" void kernel_launch(void* const* d_in, const int* in_sizes, int n_in,
                              void* d_out, int out_size, void* d_ws, size_t ws_size,
                              hipStream_t stream) {
  const float* q   = (const float*)d_in[0];
  const float* k   = (const float*)d_in[1];
  const float* v   = (const float*)d_in[2];
  const float* wq  = (const float*)d_in[3];
  const float* wk  = (const float*)d_in[4];
  const float* wv  = (const float*)d_in[5];
  const float* wfc = (const float*)d_in[6];

  const size_t NX = (size_t)BATCH * S_LEN * DMODEL;  // 4,194,304
  const size_t NW = (size_t)DMODEL * DMODEL;         // 1,048,576

  __bf16* ws  = (__bf16*)d_ws;
  __bf16* xq  = ws;            // converted inputs
  __bf16* xk  = xq  + NX;
  __bf16* xv  = xk  + NX;
  __bf16* bwq = xv  + NX;      // converted weights
  __bf16* bwk = bwq + NW;
  __bf16* bwv = bwk + NW;
  __bf16* bwfc= bwv + NW;
  __bf16* qp  = bwfc + NW;     // (B,H,S,dh) projections
  __bf16* kp  = qp  + NX;
  __bf16* vp  = kp  + NX;
  __bf16* ao  = vp  + NX;      // (B*S, D) attention output

  // 1) convert to bf16
  int gx = (int)(NX / 4 / 256), gw = (int)(NW / 4 / 256);
  cvt_f32_to_bf16<<<gx, 256, 0, stream>>>((const float4*)q,   xq,  (int)(NX / 4));
  cvt_f32_to_bf16<<<gx, 256, 0, stream>>>((const float4*)k,   xk,  (int)(NX / 4));
  cvt_f32_to_bf16<<<gx, 256, 0, stream>>>((const float4*)v,   xv,  (int)(NX / 4));
  cvt_f32_to_bf16<<<gw, 256, 0, stream>>>((const float4*)wq,  bwq, (int)(NW / 4));
  cvt_f32_to_bf16<<<gw, 256, 0, stream>>>((const float4*)wk,  bwk, (int)(NW / 4));
  cvt_f32_to_bf16<<<gw, 256, 0, stream>>>((const float4*)wv,  bwv, (int)(NW / 4));
  cvt_f32_to_bf16<<<gw, 256, 0, stream>>>((const float4*)wfc, bwfc,(int)(NW / 4));

  // 2) projections: (4096 x 1024) @ W^T, output in (B,H,S,dh)
  dim3 gg(4096 / 128, DMODEL / 64);
  gemm_xwt<<<gg, 256, 0, stream>>>(xq, bwq, qp, 0);
  gemm_xwt<<<gg, 256, 0, stream>>>(xk, bwk, kp, 0);
  gemm_xwt<<<gg, 256, 0, stream>>>(xv, bwv, vp, 0);

  // 3) attention (TDM-fed)
  attn_fwd<<<dim3(S_LEN / 128, BATCH * NHEAD), 256, 0, stream>>>(qp, kp, vp, ao);

  // 4) final projection -> f32 d_out
  gemm_xwt<<<gg, 256, 0, stream>>>(ao, bwfc, d_out, 1);
}
